// MultiHeadAttention_18614388261063
// MI455X (gfx1250) — matrix operand order repaired and checked by reading, not compiled, not run on hardware
//
#include <hip/hip_runtime.h>

// ---------------- problem constants ----------------
#define BB 4
#define TT 2048
#define DD 1024
#define HH 16
#define DKK 64
#define SCALE 0.125f            // 1/sqrt(64)
#define NEG_BIG (-1e30f)

typedef __bf16 v16bf __attribute__((ext_vector_type(16)));
typedef __bf16 v8bf  __attribute__((ext_vector_type(8)));
typedef float  v8f   __attribute__((ext_vector_type(8)));
typedef float  v4f   __attribute__((ext_vector_type(4)));

union BF16Frag { v16bf v; v8bf h[2]; };

static __device__ __forceinline__ v8f wmma_bf16(v16bf a, v16bf b, v8f c) {
    // emits v_wmma_f32_16x16x32_bf16
    return __builtin_amdgcn_wmma_f32_16x16x32_bf16(false, a, false, b, (short)0, c, false, false);
}

// Async DMA of one 16-byte chunk from global memory into LDS (ASYNCcnt-tracked).
static __device__ __forceinline__ void async_ld_b128(unsigned lds_off, const void* gaddr) {
    asm volatile("global_load_async_to_lds_b128 %0, %1, off"
                 :: "v"(lds_off), "v"((unsigned long long)(size_t)gaddr)
                 : "memory");
}

// ---------------- f32 -> bf16 convert ----------------
__global__ void cvt_f32_bf16(const float* __restrict__ in, __bf16* __restrict__ out, int n) {
    int i4 = (blockIdx.x * blockDim.x + threadIdx.x) * 4;
    if (i4 < n) {
        float4 f = *(const float4*)(in + i4);
        out[i4 + 0] = (__bf16)f.x;
        out[i4 + 1] = (__bf16)f.y;
        out[i4 + 2] = (__bf16)f.z;
        out[i4 + 3] = (__bf16)f.w;
    }
}

// ---------------- WMMA GEMM: C[M,1024] = A[M,1024] @ W[1024,1024]^T + bias ----------------
// W tile (shared by all 8 waves) staged through double-buffered LDS with
// global_load_async_to_lds_b128. Each wave owns TWO 16-row M tiles (8 WMMAs per k-step)
// to amortize the barriers and double the reuse of each staged LDS tile.
// mode 0: bf16 out, split-head [B,H,T,DK] ; mode 1: f32 out [M,1024] ; mode 2: bf16 [B,H,DK,T]
__global__ __launch_bounds__(256)
void gemm_bf16_wmma(const __bf16* __restrict__ A, const __bf16* __restrict__ W,
                    const float* __restrict__ bias, void* __restrict__ out, int mode) {
    __shared__ __bf16 bstage[2][64 * 32];   // 2 x 4KB: [n_local 0..63][k_local 0..31]

    const int tid  = threadIdx.x;
    const int lane = tid & 31;
    const int wid  = tid >> 5;
    const int lanelo = lane & 15;
    const int m0 = blockIdx.y * 256 + wid * 16;   // tile 0 rows; tile 1 = +128
    const int n0 = blockIdx.x * 64;

    const int alo = (lane < 16) ? 0 : 8;    // A frag K sub-offset
    const int blo = (lane < 16) ? 0 : 8;    // B frag K sub-offset
    const __bf16* arow0 = A + (size_t)(m0 + lanelo) * DD;
    const __bf16* arow1 = A + (size_t)(m0 + 128 + lanelo) * DD;

    // staging decomposition: 256 threads x 16B = 4KB tile (64 rows x 32 k of bf16)
    const int srow = tid >> 2, schunk = tid & 3;
    const __bf16* wstage = W + (size_t)(n0 + srow) * DD + schunk * 8;
    const unsigned lds_base0 = (unsigned)(size_t)&bstage[0][srow * 32 + schunk * 8];
    const unsigned lds_base1 = (unsigned)(size_t)&bstage[1][srow * 32 + schunk * 8];

    v8f acc[2][4];
#pragma unroll
    for (int t = 0; t < 2; ++t)
#pragma unroll
        for (int nn = 0; nn < 4; ++nn) acc[t][nn] = (v8f)0.0f;

    // prime the async pipeline: tiles k=0 and k=32
    async_ld_b128(lds_base0, wstage + 0);
    async_ld_b128(lds_base1, wstage + 32);

    BF16Frag a_cur[2];
    a_cur[0].h[0] = *(const v8bf*)(arow0 + alo);
    a_cur[0].h[1] = *(const v8bf*)(arow0 + alo + 16);
    a_cur[1].h[0] = *(const v8bf*)(arow1 + alo);
    a_cur[1].h[1] = *(const v8bf*)(arow1 + alo + 16);

    for (int it = 0; it < 32; ++it) {
        const int k0  = it * 32;
        const int cur = it & 1;

        // oldest outstanding tile (this iteration's buffer) is complete
        asm volatile("s_wait_asynccnt 0x1" ::: "memory");
        __syncthreads();

        // register double-buffer next A fragments (clamped dummy on last iter)
        const int kn_a = (k0 + 32) & (DD - 1);
        BF16Frag a_next[2];
        a_next[0].h[0] = *(const v8bf*)(arow0 + kn_a + alo);
        a_next[0].h[1] = *(const v8bf*)(arow0 + kn_a + alo + 16);
        a_next[1].h[0] = *(const v8bf*)(arow1 + kn_a + alo);
        a_next[1].h[1] = *(const v8bf*)(arow1 + kn_a + alo + 16);

        // issue ALL B-fragment ds_loads first -> partial dscnt waits, LDS/XDL co-execution
        BF16Frag ub[4];
#pragma unroll
        for (int nn = 0; nn < 4; ++nn) {
            const __bf16* lp = &bstage[cur][(nn * 16 + lanelo) * 32 + blo];
            ub[nn].h[0] = *(const v8bf*)(lp);
            ub[nn].h[1] = *(const v8bf*)(lp + 16);
        }
#pragma unroll
        for (int nn = 0; nn < 4; ++nn) {
            acc[0][nn] = wmma_bf16(a_cur[0].v, ub[nn].v, acc[0][nn]);
            acc[1][nn] = wmma_bf16(a_cur[1].v, ub[nn].v, acc[1][nn]);
        }
        a_cur[0] = a_next[0];
        a_cur[1] = a_next[1];

        __syncthreads();   // all waves done reading bstage[cur] before it is overwritten

        // stage tile k0+64 into the just-consumed buffer (clamped dummy near the end
        // keeps ASYNCcnt accounting uniform so the <=1 wait above stays correct)
        const int kn_b = (k0 + 64) & (DD - 1);
        async_ld_b128(cur ? lds_base1 : lds_base0, wstage + kn_b);
    }

    const int hi8 = (lane >> 4) << 3;
#pragma unroll
    for (int t = 0; t < 2; ++t) {
#pragma unroll
        for (int nn = 0; nn < 4; ++nn) {
            const int n = n0 + nn * 16 + lanelo;
            const float bv = bias[n];
#pragma unroll
            for (int r = 0; r < 8; ++r) {
                const int m = m0 + t * 128 + r + hi8;
                const float val = acc[t][nn][r] + bv;
                if (mode == 1) {
                    ((float*)out)[(size_t)m * DD + n] = val;
                } else {
                    const int b = m >> 11, tt = m & (TT - 1);
                    const int h = n >> 6,  dk = n & (DKK - 1);
                    size_t idx;
                    if (mode == 0) idx = ((size_t)(b * HH + h) * TT + tt) * DKK + dk;
                    else           idx = ((size_t)(b * HH + h) * DKK + dk) * TT + tt;
                    ((__bf16*)out)[idx] = (__bf16)val;
                }
            }
        }
    }
}

// ---------------- causal attention: softmax(QK^T/s) -> attn (f32) and O = P@V (bf16) --------
// Qp,Kp: [B,H,T,64] bf16 ; Vt: [B,H,64,T] bf16 ; attn: [B,H,T,T] f32 ; Ob: [B*T, 1024] bf16
__global__ __launch_bounds__(256)
void attn_wmma(const __bf16* __restrict__ Qp, const __bf16* __restrict__ Kp,
               const __bf16* __restrict__ Vt, float* __restrict__ attn,
               __bf16* __restrict__ Ob) {
    __shared__ __bf16 plds[8][16 * 32];   // per-wave P staging (16 rows x 32 keys)

    const int lane = threadIdx.x & 31;
    const int wid  = threadIdx.x >> 5;
    const int lanelo = lane & 15;
    const int hi8 = (lane >> 4) << 3;
    const int bh = blockIdx.y;                 // 0..63
    const int q0 = blockIdx.x * 128 + wid * 16;

    const __bf16* Qbh = Qp + (size_t)bh * TT * DKK;
    const __bf16* Kbh = Kp + (size_t)bh * TT * DKK;
    const __bf16* Vbh = Vt + (size_t)bh * DKK * TT;
    float* attnbh = attn + (size_t)bh * TT * TT;

    // ---- load Q A-fragments (16 rows x 64 dk, two K=32 halves) ----
    const int alo = (lane < 16) ? 0 : 8;
    const int blo = (lane < 16) ? 0 : 8;
    const __bf16* qp = Qbh + (size_t)(q0 + lanelo) * DKK;
    BF16Frag aq0, aq1;
    aq0.h[0] = *(const v8bf*)(qp + alo);
    aq0.h[1] = *(const v8bf*)(qp + alo + 16);
    aq1.h[0] = *(const v8bf*)(qp + 32 + alo);
    aq1.h[1] = *(const v8bf*)(qp + 32 + alo + 16);

    const int jmax = (q0 + 15) >> 5;           // last 32-key chunk touching the causal region

    // ================= pass 1: row max / row sum =================
    float m[8], l[8];
#pragma unroll
    for (int r = 0; r < 8; ++r) { m[r] = NEG_BIG; l[r] = 0.0f; }

    for (int j = 0; j <= jmax; ++j) {
        // batch all 8 K-fragment loads (both 16-key subtiles), then 4 WMMAs
        BF16Frag kb[2][2];
#pragma unroll
        for (int s = 0; s < 2; ++s) {
            const __bf16* kp = Kbh + (size_t)(j * 32 + s * 16 + lanelo) * DKK;
            kb[s][0].h[0] = *(const v8bf*)(kp + blo);
            kb[s][0].h[1] = *(const v8bf*)(kp + blo + 16);
            kb[s][1].h[0] = *(const v8bf*)(kp + 32 + blo);
            kb[s][1].h[1] = *(const v8bf*)(kp + 32 + blo + 16);
        }
        v8f s0 = (v8f)0.0f, s1 = (v8f)0.0f;
        s0 = wmma_bf16(aq0.v, kb[0][0].v, s0);
        s0 = wmma_bf16(aq1.v, kb[0][1].v, s0);
        s1 = wmma_bf16(aq0.v, kb[1][0].v, s1);
        s1 = wmma_bf16(aq1.v, kb[1][1].v, s1);
#pragma unroll
        for (int r = 0; r < 8; ++r) {
            const int tq = q0 + r + hi8;
            const int tk0 = j * 32 + lanelo;
            const int tk1 = tk0 + 16;
            float v0 = (tk0 > tq) ? NEG_BIG : s0[r] * SCALE;
            float v1 = (tk1 > tq) ? NEG_BIG : s1[r] * SCALE;
            // row max over 32 cols (stays within each 16-lane half)
            float tm = fmaxf(v0, v1);
            for (int off = 8; off > 0; off >>= 1) tm = fmaxf(tm, __shfl_xor(tm, off, 32));
            const float mnew = fmaxf(m[r], tm);
            float es = __expf(v0 - mnew) + __expf(v1 - mnew);
            for (int off = 8; off > 0; off >>= 1) es += __shfl_xor(es, off, 32);
            l[r] = l[r] * __expf(m[r] - mnew) + es;
            m[r] = mnew;
        }
    }
    float linv[8];
#pragma unroll
    for (int r = 0; r < 8; ++r) linv[r] = 1.0f / l[r];

    // ================= pass 2: write attn, accumulate O = P @ V =================
    v8f oacc[4];
#pragma unroll
    for (int nn = 0; nn < 4; ++nn) oacc[nn] = (v8f)0.0f;

    for (int j = 0; j <= jmax; ++j) {
        BF16Frag kb[2][2];
#pragma unroll
        for (int s = 0; s < 2; ++s) {
            const __bf16* kp = Kbh + (size_t)(j * 32 + s * 16 + lanelo) * DKK;
            kb[s][0].h[0] = *(const v8bf*)(kp + blo);
            kb[s][0].h[1] = *(const v8bf*)(kp + blo + 16);
            kb[s][1].h[0] = *(const v8bf*)(kp + 32 + blo);
            kb[s][1].h[1] = *(const v8bf*)(kp + 32 + blo + 16);
        }
#pragma unroll
        for (int s = 0; s < 2; ++s) {
            v8f sa = (v8f)0.0f;
            sa = wmma_bf16(aq0.v, kb[s][0].v, sa);
            sa = wmma_bf16(aq1.v, kb[s][1].v, sa);
#pragma unroll
            for (int r = 0; r < 8; ++r) {
                const int tq = q0 + r + hi8;
                const int tk = j * 32 + s * 16 + lanelo;
                const float v = (tk > tq) ? NEG_BIG : sa[r] * SCALE;
                const float p = __expf(v - m[r]) * linv[r];
                // streaming 1GB output: non-temporal so K/V stay resident in the 192MB L2
                __builtin_nontemporal_store(p, &attnbh[(size_t)tq * TT + tk]);
                plds[wid][(r + hi8) * 32 + s * 16 + lanelo] = (__bf16)p;
            }
        }
        asm volatile("s_wait_dscnt 0" ::: "memory");   // order per-wave DS store -> DS load

        // P as A-fragment (16 rows x 32 keys) from LDS
        BF16Frag pa;
        const __bf16* lp = &plds[wid][lanelo * 32 + alo];
        pa.h[0] = *(const v8bf*)(lp);
        pa.h[1] = *(const v8bf*)(lp + 16);

        // batch all 8 V-fragment loads, then 4 WMMAs
        BF16Frag vb[4];
#pragma unroll
        for (int nn = 0; nn < 4; ++nn) {
            const __bf16* vp = Vbh + (size_t)(nn * 16 + lanelo) * TT + j * 32 + blo;
            vb[nn].h[0] = *(const v8bf*)(vp);
            vb[nn].h[1] = *(const v8bf*)(vp + 16);
        }
#pragma unroll
        for (int nn = 0; nn < 4; ++nn)
            oacc[nn] = wmma_bf16(pa.v, vb[nn].v, oacc[nn]);
    }

    // ---- zero-fill the non-causal part of attn (non-temporal float4) ----
    const int kz = (jmax + 1) * 32;
    const v4f z4 = (v4f)0.0f;
    for (int r16 = 0; r16 < 16; ++r16) {
        float* rowp = attnbh + (size_t)(q0 + r16) * TT;
        for (int c = kz + lane * 4; c < TT; c += 128)
            __builtin_nontemporal_store(z4, (v4f*)(rowp + c));
    }

    // ---- write O (bf16) in [B*T, D] layout for the output projection ----
    const int b = bh >> 4, h = bh & 15;
#pragma unroll
    for (int nn = 0; nn < 4; ++nn) {
#pragma unroll
        for (int r = 0; r < 8; ++r) {
            const int tq = q0 + r + hi8;
            const int col = h * DKK + nn * 16 + lanelo;
            Ob[(size_t)(b * TT + tq) * DD + col] = (__bf16)oacc[nn][r];
        }
    }
}

// ---------------- host-side orchestration ----------------
extern "C" void kernel_launch(void* const* d_in, const int* in_sizes, int n_in,
                              void* d_out, int out_size, void* d_ws, size_t ws_size,
                              hipStream_t stream) {
    (void)in_sizes; (void)n_in; (void)out_size; (void)ws_size;
    const float* q  = (const float*)d_in[0];
    const float* k  = (const float*)d_in[1];
    const float* v  = (const float*)d_in[2];
    // d_in[3] = mask (causal, computed analytically)
    const float* wq = (const float*)d_in[4];
    const float* bq = (const float*)d_in[5];
    const float* wk = (const float*)d_in[6];
    const float* bk = (const float*)d_in[7];
    const float* wv = (const float*)d_in[8];
    const float* bv = (const float*)d_in[9];
    const float* wo = (const float*)d_in[10];
    const float* bo = (const float*)d_in[11];

    const size_t BTD = (size_t)BB * TT * DD;      // 8,388,608
    const size_t DxD = (size_t)DD * DD;           // 1,048,576

    // workspace carve-up (bf16 elements), ~126 MB total
    unsigned short* w = (unsigned short*)d_ws;
    unsigned short* qb  = w;            unsigned short* kb  = qb  + BTD;
    unsigned short* vb  = kb  + BTD;    unsigned short* wqb = vb  + BTD;
    unsigned short* wkb = wqb + DxD;    unsigned short* wvb = wkb + DxD;
    unsigned short* wob = wvb + DxD;    unsigned short* Qp  = wob + DxD;
    unsigned short* Kp  = Qp  + BTD;    unsigned short* Vt  = Kp  + BTD;
    unsigned short* Ob  = Vt  + BTD;

    float* out_f = (float*)d_out;                 // [B,T,D]
    float* attn  = out_f + BTD;                   // [B,H,T,T]

    const int cvtThreads = 256;
    auto cvt = [&](const float* src, unsigned short* dst, size_t n) {
        int blocks = (int)((n / 4 + cvtThreads - 1) / cvtThreads);
        cvt_f32_bf16<<<blocks, cvtThreads, 0, stream>>>(src, (__bf16*)dst, (int)n);
    };
    cvt(q, qb, BTD); cvt(k, kb, BTD); cvt(v, vb, BTD);
    cvt(wq, wqb, DxD); cvt(wk, wkb, DxD); cvt(wv, wvb, DxD); cvt(wo, wob, DxD);

    const dim3 gGrid(DD / 64, (BB * TT) / 256);   // (16, 32)
    gemm_bf16_wmma<<<gGrid, 256, 0, stream>>>((const __bf16*)qb, (const __bf16*)wqb, bq, Qp, 0);
    gemm_bf16_wmma<<<gGrid, 256, 0, stream>>>((const __bf16*)kb, (const __bf16*)wkb, bk, Kp, 0);
    gemm_bf16_wmma<<<gGrid, 256, 0, stream>>>((const __bf16*)vb, (const __bf16*)wvb, bv, Vt, 2);

    const dim3 aGrid(TT / 128, BB * HH);          // (16, 64)
    attn_wmma<<<aGrid, 256, 0, stream>>>((const __bf16*)Qp, (const __bf16*)Kp,
                                         (const __bf16*)Vt, attn, (__bf16*)Ob);

    gemm_bf16_wmma<<<gGrid, 256, 0, stream>>>((const __bf16*)Ob, (const __bf16*)wob, bo, out_f, 1);
}
